// DisentangledVAE_75711683494094
// MI455X (gfx1250) — matrix operand order repaired and checked by reading, not compiled, not run on hardware
//
#include <hip/hip_runtime.h>
#include <hip/hip_bf16.h>

#if defined(__gfx1250__) && __has_builtin(__builtin_amdgcn_global_load_async_to_lds_b128)
#define ASYNC_LDS 1
#else
#define ASYNC_LDS 0
#endif

#define AS1 __attribute__((address_space(1)))
#define AS3 __attribute__((address_space(3)))

namespace {

constexpr int Zc = 256, Tc = 16, Bc = 1024;
constexpr int Rc = Bc * Tc;          // 16384 flattened rows
constexpr int NFc = 512, Hc = 512;
constexpr float LSP = 0.9189385332046727f;   // 0.5*log(2*pi)

typedef __attribute__((ext_vector_type(16))) __bf16 v16bf;
typedef __attribute__((ext_vector_type(8)))  float  v8f;
typedef __attribute__((ext_vector_type(4)))  int    i4;
typedef unsigned short u16;

__device__ __forceinline__ u16 f2bf(float f) {
  unsigned u = __float_as_uint(f);
  u += 0x7FFFu + ((u >> 16) & 1u);          // round-to-nearest-even
  return (u16)(u >> 16);
}
__device__ __forceinline__ float sigm(float x) { return 1.f / (1.f + __expf(-x)); }

union FragAB { uint4 q[2]; v16bf v; };

// async global->LDS b128 copy (CDNA5 path, ASYNCcnt-tracked)
__device__ __forceinline__ void async_b128(const u16* g, u16* l) {
#if ASYNC_LDS
  __builtin_amdgcn_global_load_async_to_lds_b128(
      (AS1 i4*)(void*)g, (AS3 i4*)(void*)l, 0, 0);
#else
  (void)g; (void)l;
#endif
}
__device__ __forceinline__ void wait_async0() {
#if ASYNC_LDS
#if __has_builtin(__builtin_amdgcn_s_wait_asynccnt)
  __builtin_amdgcn_s_wait_asynccnt(0);
#else
  asm volatile("s_wait_asynccnt 0" ::: "memory");
#endif
#endif
}

// ---------------- elementwise / utility kernels ----------------

__global__ void k_f32_to_bf16(const float* __restrict__ in, u16* __restrict__ out, int n) {
  for (int i = blockIdx.x * blockDim.x + threadIdx.x; i < n; i += blockDim.x * gridDim.x)
    out[i] = f2bf(in[i]);
}
__global__ void k_zero_f32(float* p, int n) {
  for (int i = blockIdx.x * blockDim.x + threadIdx.x; i < n; i += blockDim.x * gridDim.x) p[i] = 0.f;
}
__global__ void k_zero_u16(u16* p, int n) {
  for (int i = blockIdx.x * blockDim.x + threadIdx.x; i < n; i += blockDim.x * gridDim.x) p[i] = 0;
}

// nf_input [R, 2Z] -> p (bf16) and z = posterior latents (f32)
__global__ void k_extract(const float* __restrict__ nf, float* __restrict__ z, u16* __restrict__ pbf) {
  int row = blockIdx.x, zi = threadIdx.x;
  size_t src = (size_t)row * (2 * Zc);
  z[(size_t)row * Zc + zi]   = nf[src + Zc + zi];
  pbf[(size_t)row * Zc + zi] = f2bf(nf[src + zi]);
}

// build coupling-layer MLP input [R, 512] in bf16.  lt: 0='c', 1='a'(odd masked), 2='b'(even masked)
__global__ void k_prep(const float* __restrict__ z, const u16* __restrict__ pbf,
                       u16* __restrict__ inp, int lt) {
  int row = blockIdx.x, zi = threadIdx.x;
  float v;
  if (lt == 0) {                       // z_shift within each sequence of T frames
    int t = row & (Tc - 1);
    v = (t == 0) ? 0.f : z[(size_t)(row - 1) * Zc + zi];
  } else {
    bool keep = (lt == 1) ? ((zi & 1) == 1) : ((zi & 1) == 0);
    v = keep ? z[(size_t)row * Zc + zi] : 0.f;
  }
  inp[(size_t)row * 512 + zi]      = f2bf(v);
  inp[(size_t)row * 512 + Zc + zi] = pbf[(size_t)row * Zc + zi];
}

__device__ __forceinline__ float block_reduce_256(float v) {
  __shared__ float red[256];
  red[threadIdx.x] = v;
  __syncthreads();
  for (int s = 128; s > 0; s >>= 1) {
    if ((int)threadIdx.x < s) red[threadIdx.x] += red[threadIdx.x + s];
    __syncthreads();
  }
  return red[0];
}

// z = (z - t)*exp(-s) on unmasked lanes; ldj[row] -= sum(s over unmasked)
__global__ void k_update(float* __restrict__ z, const float* __restrict__ sO,
                         const float* __restrict__ tO, float* __restrict__ ldj, int lt) {
  int row = blockIdx.x, zi = threadIdx.x;
  size_t idx = (size_t)row * Zc + zi;
  bool upd = (lt == 0) || ((lt == 1) ? ((zi & 1) == 0) : ((zi & 1) == 1));
  float sv = 0.f;
  if (upd) {
    float s = sO[idx], t = tO[idx];
    z[idx] = (z[idx] - t) * __expf(-s);
    sv = s;
  }
  float tot = block_reduce_256(sv);
  if (threadIdx.x == 0) ldj[row] -= tot;
}

// per-row sum of logN(z3)
__global__ void k_zterm(const float* __restrict__ z, float* __restrict__ part) {
  int row = blockIdx.x, zi = threadIdx.x;
  float v = z[(size_t)row * Zc + zi];
  float tot = block_reduce_256(-0.5f * v * v - LSP);
  if (threadIdx.x == 0) part[row] = tot;
}

// torch LSTMCell gate order i,f,g,o ; gates [B, 4H]
__global__ void k_cell(const float* __restrict__ g, float* __restrict__ c, u16* __restrict__ hbf) {
  int b = blockIdx.x;
  int j = blockIdx.y * 256 + threadIdx.x;      // 0..H-1
  size_t g0 = (size_t)b * (4 * Hc);
  float ig = sigm(g[g0 + j]);
  float fg = sigm(g[g0 + Hc + j]);
  float gg = tanhf(g[g0 + 2 * Hc + j]);
  float og = sigm(g[g0 + 3 * Hc + j]);
  size_t ci = (size_t)b * Hc + j;
  float c2 = fg * c[ci] + ig * gg;
  c[ci] = c2;
  hbf[ci] = f2bf(og * tanhf(c2));
}

// samp = (z_t - mean)*exp(-logvar); partial[t*B+b] = sum(logN(samp) - logvar)
__global__ void k_prior(const float* __restrict__ z3, const float* __restrict__ mean,
                        const float* __restrict__ lvar, float* __restrict__ part, int t) {
  int b = blockIdx.x, zi = threadIdx.x;
  float zv = z3[((size_t)b * Tc + t) * Zc + zi];
  size_t i = (size_t)b * Zc + zi;
  float lv = lvar[i];
  float samp = (zv - mean[i]) * __expf(-lv);
  float tot = block_reduce_256(-0.5f * samp * samp - LSP - lv);
  if (threadIdx.x == 0) part[t * Bc + b] = tot;
}

__global__ void k_final(const float* __restrict__ pPrior, const float* __restrict__ pZ,
                        const float* __restrict__ ldj, float* __restrict__ out) {
  float s1 = 0.f, s2 = 0.f;
  for (int i = threadIdx.x; i < Rc; i += 256) { s1 += pPrior[i]; s2 += pZ[i] + ldj[i]; }
  float tot = block_reduce_256(s1 * 0.0025f + s2);
  if (threadIdx.x == 0) out[0] = -tot / (float)Bc;
}

// ---------------- WMMA GEMM ----------------
// C[M,N] = act( A[M,K]@W[N,K]^T + (A2[M,K2]@W2[N,K2]^T)? + bias + bias2? )
// BM=128, BN=64; 8 wave32 waves, each 32x32 via 2x2 v_wmma_f32_16x16x32_bf16.
// Double-buffered LDS; global->LDS via async b128 copies (ASYNCcnt) when available.
__global__ __launch_bounds__(256)
void k_wmma_gemm(const u16* __restrict__ A, int lda,
                 const u16* __restrict__ A2, int lda2,
                 const u16* __restrict__ W, const u16* __restrict__ W2,
                 const float* __restrict__ bias, const float* __restrict__ bias2,
                 float* __restrict__ outF, u16* __restrict__ outH,
                 int ldo, int K, int K2, int act) {
  constexpr int BM = 128, BN = 64, BK = 32, LS = 40;   // LS: padded LDS row stride (u16)
  __shared__ u16 sA[2][BM][LS];
  __shared__ u16 sB[2][BN][LS];

  const int tid  = threadIdx.x;
  const int wave = tid >> 5, lane = tid & 31;
  const int wm = (wave & 3) * 32, wn = (wave >> 2) * 32;
  const int mlo = lane & 15, sel = lane >> 4;
  const int rowBase = blockIdx.y * BM;
  const int colBase = blockIdx.x * BN;

  // static tile-staging assignment: thread owns A rows (tid>>2, tid>>2+64) and B row tid>>2,
  // 8 bf16 (one b128) each at column (tid&3)*8
  const int sr = tid >> 2, sk = (tid & 3) << 3;

  v8f acc[2][2] = {};

  // stage one 32-wide K slab of A (128x32) + B (64x32) into LDS buffer `buf`
  auto stage = [&](int buf, const u16* Ap, int ldap, const u16* Wp, int Kp, int kb) {
    const u16* ga0 = Ap + (size_t)(rowBase + sr) * ldap + kb + sk;
    const u16* ga1 = Ap + (size_t)(rowBase + sr + 64) * ldap + kb + sk;
    const u16* gb  = Wp + (size_t)(colBase + sr) * Kp + kb + sk;
#if ASYNC_LDS
    async_b128(ga0, &sA[buf][sr][sk]);
    async_b128(ga1, &sA[buf][sr + 64][sk]);
    async_b128(gb,  &sB[buf][sr][sk]);
#else
    uint4 a0 = *(const uint4*)ga0;
    uint4 a1 = *(const uint4*)ga1;
    uint4 b0 = *(const uint4*)gb;
    *(uint4*)&sA[buf][sr][sk]      = a0;
    *(uint4*)&sA[buf][sr + 64][sk] = a1;
    *(uint4*)&sB[buf][sr][sk]      = b0;
#endif
    if (kb + BK < Kp) {          // warm L2 for the slab after this one
      __builtin_prefetch(ga0 + BK, 0, 1);
      __builtin_prefetch(gb + BK, 0, 1);
    }
  };

  auto compute = [&](int buf) {
    FragAB fa[2], fb[2];
#pragma unroll
    for (int r = 0; r < 2; ++r) {   // A 16x32: row = lane&15 ; K halves by lane>>4
      const u16* p = &sA[buf][wm + r * 16 + mlo][sel * 8];
      fa[r].q[0] = *(const uint4*)p;          // K = sel*8 .. +7
      fa[r].q[1] = *(const uint4*)(p + 16);   // K = sel*8+16 .. +23
    }
#pragma unroll
    for (int c = 0; c < 2; ++c) {   // B 32x16: col = lane&15 ; K half by lane>>4
      const u16* p = &sB[buf][wn + c * 16 + mlo][sel * 8];
      fb[c].q[0] = *(const uint4*)p;          // K = sel*8 .. +7
      fb[c].q[1] = *(const uint4*)(p + 16);   // K = sel*8+16 .. +23
    }
#pragma unroll
    for (int r = 0; r < 2; ++r)
#pragma unroll
      for (int c = 0; c < 2; ++c)
        acc[r][c] = __builtin_amdgcn_wmma_f32_16x16x32_bf16(
            false, fa[r].v, false, fb[c].v, (short)0, acc[r][c], false, false);
  };

  for (int pass = 0; pass < 2; ++pass) {
    const u16* Ap = pass ? A2 : A;
    const u16* Wp = pass ? W2 : W;
    const int Kp   = pass ? K2 : K;
    const int ldap = pass ? lda2 : lda;
    if (Kp == 0) continue;

    stage(0, Ap, ldap, Wp, Kp, 0);
    const int nk = Kp / BK;
    for (int ik = 0; ik < nk; ++ik) {
      const int cur = ik & 1;
      wait_async0();                 // my async copies into `cur` (and last `cur^1`) landed
      __syncthreads();               // everyone's copies visible; prior reads of cur^1 done
      if (ik + 1 < nk) stage(cur ^ 1, Ap, ldap, Wp, Kp, (ik + 1) * BK);
      compute(cur);
    }
    __syncthreads();                 // protect buffer 0 before next pass restages it
  }

  // epilogue: C/D layout — N = lane&15 ; M = vgpr + 8*(lane>>4)
#pragma unroll
  for (int c = 0; c < 2; ++c) {
    int gcol = colBase + wn + c * 16 + mlo;
    float bs = bias ? bias[gcol] : 0.f;
    if (bias2) bs += bias2[gcol];
#pragma unroll
    for (int r = 0; r < 2; ++r) {
      int growb = rowBase + wm + r * 16 + sel * 8;
#pragma unroll
      for (int v = 0; v < 8; ++v) {
        float x = acc[r][c][v] + bs;
        if (act == 1)      x = (x > 0.f) ? x : 0.01f * x;   // leaky_relu(0.01)
        else if (act == 2) x = tanhf(x);
        size_t o = (size_t)(growb + v) * ldo + gcol;
        if (outF) outF[o] = x;
        if (outH) outH[o] = f2bf(x);
      }
    }
  }
}

// ---------------- host helpers ----------------

static inline void conv(hipStream_t st, const void* src, void* dst, int n) {
  int blocks = (n + 255) / 256; if (blocks > 8192) blocks = 8192;
  k_f32_to_bf16<<<blocks, 256, 0, st>>>((const float*)src, (u16*)dst, n);
}
static inline void zf(hipStream_t st, float* p, int n) {
  int blocks = (n + 255) / 256; if (blocks > 8192) blocks = 8192;
  k_zero_f32<<<blocks, 256, 0, st>>>(p, n);
}
static inline void zu(hipStream_t st, u16* p, int n) {
  int blocks = (n + 255) / 256; if (blocks > 8192) blocks = 8192;
  k_zero_u16<<<blocks, 256, 0, st>>>(p, n);
}
static void gemm(hipStream_t st, int M, int N, int K,
                 const void* A, int lda, const void* W, const float* bias,
                 float* outF, void* outH, int act,
                 const void* A2 = nullptr, int lda2 = 0, const void* W2 = nullptr,
                 int K2 = 0, const float* bias2 = nullptr) {
  dim3 grid(N / 64, M / 128);
  k_wmma_gemm<<<grid, 256, 0, st>>>((const u16*)A, lda, (const u16*)A2, lda2,
                                    (const u16*)W, (const u16*)W2, bias, bias2,
                                    outF, (u16*)outH, N, K, K2, act);
}

} // namespace

extern "C" void kernel_launch(void* const* d_in, const int* in_sizes, int n_in,
                              void* d_out, int out_size, void* d_ws, size_t ws_size,
                              hipStream_t stream) {
  (void)in_sizes; (void)n_in; (void)out_size; (void)ws_size;

  const float* nf   = (const float*)d_in[0];
  const float* sW1  = (const float*)d_in[1];  const float* sb1 = (const float*)d_in[2];
  const float* sW2  = (const float*)d_in[3];  const float* sb2 = (const float*)d_in[4];
  const float* sW3  = (const float*)d_in[5];  const float* sb3 = (const float*)d_in[6];
  const float* tW1  = (const float*)d_in[7];  const float* tb1 = (const float*)d_in[8];
  const float* tW2  = (const float*)d_in[9];  const float* tb2 = (const float*)d_in[10];
  const float* tW3  = (const float*)d_in[11]; const float* tb3 = (const float*)d_in[12];
  const float* l1Wih = (const float*)d_in[13]; const float* l1Whh = (const float*)d_in[14];
  const float* l1bih = (const float*)d_in[15]; const float* l1bhh = (const float*)d_in[16];
  const float* l2Wih = (const float*)d_in[17]; const float* l2Whh = (const float*)d_in[18];
  const float* l2bih = (const float*)d_in[19]; const float* l2bhh = (const float*)d_in[20];
  const float* mW  = (const float*)d_in[21]; const float* mb  = (const float*)d_in[22];
  const float* lvW = (const float*)d_in[23]; const float* lvb = (const float*)d_in[24];

  size_t off = 0;
  auto alloc = [&](size_t bytes) -> void* {
    void* p = (char*)d_ws + off;
    off += (bytes + 255) & ~(size_t)255;
    return p;
  };
  // fp32 scratch
  float* z      = (float*)alloc((size_t)Rc * Zc * 4);
  float* sOut   = (float*)alloc((size_t)Rc * Zc * 4);
  float* tOut   = (float*)alloc((size_t)Rc * Zc * 4);
  float* ldj    = (float*)alloc((size_t)Rc * 4);
  float* gates1 = (float*)alloc((size_t)Bc * 4 * Hc * 4);
  float* gates2 = (float*)alloc((size_t)Bc * 4 * Hc * 4);
  float* c1     = (float*)alloc((size_t)Bc * Hc * 4);
  float* c2     = (float*)alloc((size_t)Bc * Hc * 4);
  float* mean   = (float*)alloc((size_t)Bc * Zc * 4);
  float* lvar   = (float*)alloc((size_t)Bc * Zc * 4);
  float* pPrior = (float*)alloc((size_t)Rc * 4);
  float* pZ     = (float*)alloc((size_t)Rc * 4);
  // bf16 scratch
  u16* pbf   = (u16*)alloc((size_t)Rc * Zc * 2);
  u16* inp   = (u16*)alloc((size_t)Rc * 512 * 2);
  u16* h1m   = (u16*)alloc((size_t)Rc * NFc * 2);
  u16* h2m   = (u16*)alloc((size_t)Rc * NFc * 2);
  u16* z3bf  = (u16*)alloc((size_t)Rc * Zc * 2);
  u16* h1s   = (u16*)alloc((size_t)Bc * Hc * 2);
  u16* h2s   = (u16*)alloc((size_t)Bc * Hc * 2);
  u16* sW1b  = (u16*)alloc((size_t)9 * NFc * 512 * 2);
  u16* sW2b  = (u16*)alloc((size_t)9 * NFc * NFc * 2);
  u16* sW3b  = (u16*)alloc((size_t)9 * Zc * NFc * 2);
  u16* tW1b  = (u16*)alloc((size_t)9 * NFc * 512 * 2);
  u16* tW2b  = (u16*)alloc((size_t)9 * NFc * NFc * 2);
  u16* tW3b  = (u16*)alloc((size_t)9 * Zc * NFc * 2);
  u16* l1Wihb = (u16*)alloc((size_t)4 * Hc * Zc * 2);
  u16* l1Whhb = (u16*)alloc((size_t)4 * Hc * Hc * 2);
  u16* l2Wihb = (u16*)alloc((size_t)4 * Hc * Hc * 2);
  u16* l2Whhb = (u16*)alloc((size_t)4 * Hc * Hc * 2);
  u16* mWb    = (u16*)alloc((size_t)Zc * Hc * 2);
  u16* lvWb   = (u16*)alloc((size_t)Zc * Hc * 2);

  // weight conversion (bf16)
  conv(stream, sW1, sW1b, 9 * NFc * 512); conv(stream, sW2, sW2b, 9 * NFc * NFc);
  conv(stream, sW3, sW3b, 9 * Zc * NFc);
  conv(stream, tW1, tW1b, 9 * NFc * 512); conv(stream, tW2, tW2b, 9 * NFc * NFc);
  conv(stream, tW3, tW3b, 9 * Zc * NFc);
  conv(stream, l1Wih, l1Wihb, 4 * Hc * Zc); conv(stream, l1Whh, l1Whhb, 4 * Hc * Hc);
  conv(stream, l2Wih, l2Wihb, 4 * Hc * Hc); conv(stream, l2Whh, l2Whhb, 4 * Hc * Hc);
  conv(stream, mW, mWb, Zc * Hc); conv(stream, lvW, lvWb, Zc * Hc);

  // state init
  k_extract<<<Rc, 256, 0, stream>>>(nf, z, pbf);
  zf(stream, ldj, Rc); zf(stream, c1, Bc * Hc); zf(stream, c2, Bc * Hc);
  zu(stream, h1s, Bc * Hc); zu(stream, h2s, Bc * Hc);

  // inverse flow: layers 8..0
  for (int i = 8; i >= 0; --i) {
    int lt = (i == 0) ? 0 : ((i & 1) ? 1 : 2);
    k_prep<<<Rc, 256, 0, stream>>>(z, pbf, inp, lt);
    const u16* w1s = sW1b + (size_t)i * NFc * 512;
    const u16* w2s = sW2b + (size_t)i * NFc * NFc;
    const u16* w3s = sW3b + (size_t)i * Zc * NFc;
    const u16* w1t = tW1b + (size_t)i * NFc * 512;
    const u16* w2t = tW2b + (size_t)i * NFc * NFc;
    const u16* w3t = tW3b + (size_t)i * Zc * NFc;
    // s-MLP (final tanh)
    gemm(stream, Rc, NFc, 512, inp, 512, w1s, sb1 + (size_t)i * NFc, nullptr, h1m, 1);
    gemm(stream, Rc, NFc, NFc, h1m, NFc, w2s, sb2 + (size_t)i * NFc, nullptr, h2m, 1);
    gemm(stream, Rc, Zc,  NFc, h2m, NFc, w3s, sb3 + (size_t)i * Zc, sOut, nullptr, 2);
    // t-MLP (linear out)
    gemm(stream, Rc, NFc, 512, inp, 512, w1t, tb1 + (size_t)i * NFc, nullptr, h1m, 1);
    gemm(stream, Rc, NFc, NFc, h1m, NFc, w2t, tb2 + (size_t)i * NFc, nullptr, h2m, 1);
    gemm(stream, Rc, Zc,  NFc, h2m, NFc, w3t, tb3 + (size_t)i * Zc, tOut, nullptr, 0);
    k_update<<<Rc, 256, 0, stream>>>(z, sOut, tOut, ldj, lt);
  }

  conv(stream, z, z3bf, Rc * Zc);          // zp = stop_gradient(z3)
  k_zterm<<<Rc, 256, 0, stream>>>(z, pZ);  // sum logN(z3)

  // LSTM prior, sequential over T
  for (int t = 0; t < Tc; ++t) {
    gemm(stream, Bc, 4 * Hc, Zc, z3bf + (size_t)t * Zc, Tc * Zc, l1Wihb, l1bih,
         gates1, nullptr, 0, h1s, Hc, l1Whhb, Hc, l1bhh);
    k_cell<<<dim3(Bc, 2), 256, 0, stream>>>(gates1, c1, h1s);
    gemm(stream, Bc, 4 * Hc, Hc, h1s, Hc, l2Wihb, l2bih,
         gates2, nullptr, 0, h2s, Hc, l2Whhb, Hc, l2bhh);
    k_cell<<<dim3(Bc, 2), 256, 0, stream>>>(gates2, c2, h2s);
    gemm(stream, Bc, Zc, Hc, h2s, Hc, mWb,  mb,  mean, nullptr, 0);
    gemm(stream, Bc, Zc, Hc, h2s, Hc, lvWb, lvb, lvar, nullptr, 0);
    k_prior<<<Bc, 256, 0, stream>>>(z, mean, lvar, pPrior, t);
  }

  k_final<<<1, 256, 0, stream>>>(pPrior, pZ, ldj, (float*)d_out);
  (void)hipMemcpyAsync((float*)d_out + 1, z, (size_t)Rc * Zc * sizeof(float),
                       hipMemcpyDeviceToDevice, stream);
}
